// ExperimentalLayer6_1563368096369
// MI455X (gfx1250) — matrix operand order repaired and checked by reading, not compiled, not run on hardware
//
#include <hip/hip_runtime.h>
#include <hip/hip_bf16.h>

typedef _Float16 half_t;
typedef __attribute__((ext_vector_type(16))) _Float16 v16h;
typedef __attribute__((ext_vector_type(8)))  _Float16 v8h;
typedef __attribute__((ext_vector_type(4)))  _Float16 v4h;
typedef __attribute__((ext_vector_type(8)))  float    v8f;
typedef __attribute__((ext_vector_type(4)))  float    v4f;
typedef __attribute__((ext_vector_type(4)))  int      v4i;

#define D_MODEL 768
#define BATCH   2
#define N_CTX   4096
#define M_TOTAL (BATCH * N_CTX)   // 8192
#define E_TOTAL (3 * D_MODEL)     // 2304

// ---------------------------------------------------------------------------
// CDNA5 async global->LDS copies (ASYNCcnt-tracked), guarded so the file
// still compiles (sync fallback) if this toolchain lacks the builtins.
// Builtin signature (from hipcc diagnostic): (v4i addrspace(1)*, v4i addrspace(3)*, imm, imm)
// ---------------------------------------------------------------------------
#if defined(__HIP_DEVICE_COMPILE__) && \
    __has_builtin(__builtin_amdgcn_global_load_async_to_lds_b128) && \
    __has_builtin(__builtin_amdgcn_s_wait_asynccnt)
#define USE_ASYNC 1
#else
#define USE_ASYNC 0
#endif

#if USE_ASYNC
#define GLB_AS __attribute__((address_space(1)))
#define LDS_AS __attribute__((address_space(3)))
__device__ __forceinline__ void async_copy16(void* lds_dst, const void* gsrc) {
  // global_load_async_to_lds_b128: per-lane 16B global -> per-lane LDS addr
  __builtin_amdgcn_global_load_async_to_lds_b128(
      (GLB_AS v4i*)gsrc, (LDS_AS v4i*)lds_dst, 0, 0);
}
#define ASYNC_WAIT(n) __builtin_amdgcn_s_wait_asynccnt(n)
#endif

// ---------------------------------------------------------------------------
// Kernel 0: fp32 -> fp16 conversion (vectorized x4)
// ---------------------------------------------------------------------------
__global__ void cvt_f32_f16(const float* __restrict__ in, half_t* __restrict__ out, int n) {
  int i = (blockIdx.x * blockDim.x + threadIdx.x) * 4;
  if (i < n) {
    v4f f = *(const v4f*)(in + i);
    *(v4h*)(out + i) = __builtin_convertvector(f, v4h);
  }
}

// ---------------------------------------------------------------------------
// Kernel 1: QKV projection GEMM
//   QKV[m][e] = sum_d Xh[m][d] * Wh[e][d]     (torch Linear: x @ W^T)
//   M=8192, N=2304, K=768. Block tile 128x128, 8 waves, wave tile 32x64.
//   Double-buffered LDS staging via async global->LDS copies.
// ---------------------------------------------------------------------------
#define GBM 128
#define GBN 128
#define GBK 32
#define GLS 48   // LDS row stride in halves (32 + 16 pad) = 96B, 16B aligned

__global__ __launch_bounds__(256)
void qkv_gemm(const half_t* __restrict__ X,   // [8192][768]
              const half_t* __restrict__ W,   // [2304][768]
              half_t* __restrict__ QKV)       // [8192][2304]
{
  __shared__ half_t As[2][GBM * GLS];
  __shared__ half_t Bs[2][GBN * GLS];

  const int tid  = threadIdx.x;
  const int wave = tid >> 5;
  const int lane = tid & 31;
  const int l16  = lane & 15;
  const int lhi  = lane >> 4;

  const int m0 = blockIdx.x * GBM;
  const int n0 = blockIdx.y * GBN;
  const int wm = (wave >> 1) * 32;   // 4 waves along M
  const int wn = (wave & 1) * 64;    // 2 waves along N

  v8f acc[2][4] = {};

  const int ldr = tid >> 1;          // 0..127 tile row
  const int ldc = (tid & 1) * 16;    // 0 or 16
  const half_t* sa_base = X + (size_t)(m0 + ldr) * D_MODEL + ldc;
  const half_t* sb_base = W + (size_t)(n0 + ldr) * D_MODEL + ldc;
  const int lofs = ldr * GLS + ldc;

#if USE_ASYNC
  // prologue: stage tile 0 into buffer 0
  async_copy16(&As[0][lofs],     sa_base);
  async_copy16(&As[0][lofs + 8], sa_base + 8);
  async_copy16(&Bs[0][lofs],     sb_base);
  async_copy16(&Bs[0][lofs + 8], sb_base + 8);
#endif

  const int NT = D_MODEL / GBK;   // 24 k-tiles
  for (int kt = 0; kt < NT; ++kt) {
    const int buf = kt & 1;
#if USE_ASYNC
    if (kt + 1 < NT) {           // prefetch next tile into alternate buffer
      const int nb = buf ^ 1;
      const int kk = (kt + 1) * GBK;
      async_copy16(&As[nb][lofs],     sa_base + kk);
      async_copy16(&As[nb][lofs + 8], sa_base + kk + 8);
      async_copy16(&Bs[nb][lofs],     sb_base + kk);
      async_copy16(&Bs[nb][lofs + 8], sb_base + kk + 8);
      ASYNC_WAIT(4);             // tile kt's 4 copies have landed
    } else {
      ASYNC_WAIT(0);
    }
    __syncthreads();
#else
    {
      const int kk = kt * GBK;
      *(v8h*)&As[buf][lofs]     = *(const v8h*)(sa_base + kk);
      *(v8h*)&As[buf][lofs + 8] = *(const v8h*)(sa_base + kk + 8);
      *(v8h*)&Bs[buf][lofs]     = *(const v8h*)(sb_base + kk);
      *(v8h*)&Bs[buf][lofs + 8] = *(const v8h*)(sb_base + kk + 8);
    }
    __syncthreads();
#endif

    union Frag { v16h v; v8h h[2]; };
    Frag af[2], bf[4];
    // A fragments (16x32): lanes 0-15 hold K 0..15, lanes 16-31 K 16..31
    #pragma unroll
    for (int rb = 0; rb < 2; ++rb) {
      const half_t* p = &As[buf][(wm + rb * 16 + l16) * GLS];
      af[rb].h[0] = *(const v8h*)(p + 16 * lhi);
      af[rb].h[1] = *(const v8h*)(p + 16 * lhi + 8);
    }
    // B fragments (32x16): lane = column, K contiguous (0-15 / 16-31 per half-wave)
    #pragma unroll
    for (int cb = 0; cb < 4; ++cb) {
      const half_t* p = &Bs[buf][(wn + cb * 16 + l16) * GLS + 16 * lhi];
      bf[cb].h[0] = *(const v8h*)p;
      bf[cb].h[1] = *(const v8h*)(p + 8);
    }
    #pragma unroll
    for (int rb = 0; rb < 2; ++rb)
      #pragma unroll
      for (int cb = 0; cb < 4; ++cb)
        acc[rb][cb] = __builtin_amdgcn_wmma_f32_16x16x32_f16(
            false, af[rb].v, false, bf[cb].v, (short)0, acc[rb][cb], false, false);
    __syncthreads();
  }

  // store C: lane L, VGPR r -> row = base + r + 8*(L>>4), col = base + (L&15)
  #pragma unroll
  for (int rb = 0; rb < 2; ++rb) {
    #pragma unroll
    for (int cb = 0; cb < 4; ++cb) {
      const int row = m0 + wm + rb * 16 + 8 * lhi;
      const int col = n0 + wn + cb * 16 + l16;
      half_t* out = QKV + (size_t)row * E_TOTAL + col;
      #pragma unroll
      for (int r = 0; r < 8; ++r)
        out[(size_t)r * E_TOTAL] = (half_t)acc[rb][cb][r];
    }
  }
}

// ---------------------------------------------------------------------------
// Kernel 2: flash attention (causal, no 1/sqrt(d) scale) + residual
//   32 queries per WG, 64-key tiles, online softmax, D=768.
//   LDS: Qs[32][776]h | KV: K[64][776]h reused as Vt[768][72]h | Ss[32][66]f
//        | Ps[32][72]h | m/l/corr[32]f   (~170 KB dynamic LDS)
// ---------------------------------------------------------------------------
#define MQ 32
#define NK 64
#define QS_STRIDE (D_MODEL + 8)   // 776
#define KS_STRIDE (D_MODEL + 8)   // 776
#define VT_STRIDE (NK + 8)        // 72
#define PS_STRIDE (NK + 8)        // 72
#define SS_STRIDE (NK + 2)        // 66
#define KV_HALVES (D_MODEL * VT_STRIDE)   // 55296 > 64*776=49664

#define ATTN_LDS_BYTES                                                  \
  ( MQ * QS_STRIDE * 2      /* Qs   */                                  \
  + KV_HALVES * 2           /* K / Vt union */                          \
  + MQ * SS_STRIDE * 4      /* Ss   */                                  \
  + MQ * PS_STRIDE * 2      /* Ps   */                                  \
  + 3 * MQ * 4 )            /* m,l,corr */

__global__ __launch_bounds__(256)
void attn_kernel(const half_t* __restrict__ QKV,   // [8192][2304] (Q|K|V)
                 const float* __restrict__ X,      // [8192][768]
                 float* __restrict__ Out)          // [8192][768]
{
  extern __shared__ char smem[];
  half_t* Qs   = (half_t*)smem;
  half_t* KV   = Qs + MQ * QS_STRIDE;
  float*  Ss   = (float*)(KV + KV_HALVES);
  half_t* Ps   = (half_t*)(Ss + MQ * SS_STRIDE);
  float*  mrow = (float*)(Ps + MQ * PS_STRIDE);
  float*  lrow = mrow + MQ;
  float*  crow = lrow + MQ;

  const int tid  = threadIdx.x;
  const int wave = tid >> 5;
  const int lane = tid & 31;
  const int l16  = lane & 15;
  const int lhi  = lane >> 4;

  const int qt = blockIdx.x;                 // 0..255
  const int b  = qt / (N_CTX / MQ);
  const int q0 = (qt % (N_CTX / MQ)) * MQ;
  const size_t rowbase = (size_t)(b * N_CTX) * E_TOTAL;
  const half_t* Kp = QKV + D_MODEL;
  const half_t* Vp = QKV + 2 * D_MODEL;

  // ---- stage Q tile (32 x 768): 8 threads/row, 96 halves each ----
  {
    const int r  = tid >> 3;
    const int c0 = (tid & 7) * 96;
    const half_t* src = QKV + rowbase + (size_t)(q0 + r) * E_TOTAL + c0;
    half_t* dst = Qs + r * QS_STRIDE + c0;
#if USE_ASYNC
    #pragma unroll
    for (int i = 0; i < 96; i += 8) async_copy16(dst + i, src + i);
#else
    #pragma unroll
    for (int i = 0; i < 96; i += 8) *(v8h*)(dst + i) = *(const v8h*)(src + i);
#endif
  }
  if (tid < MQ) { mrow[tid] = -3.0e38f; lrow[tid] = 0.0f; }

  // per-wave output accumulator: 16-row blocks x 96-col D slice
  v8f o[2][6] = {};

  const int jend = (q0 + MQ - 1) / NK;
  for (int j = 0; j <= jend; ++j) {
    const int k0 = j * NK;
    __syncthreads();   // prior-iteration Vt reads done; Qs/stats visible

    // ---- stage K tile (64 x 768): 4 threads/row, 192 halves each ----
    {
      const int r  = tid >> 2;
      const int c0 = (tid & 3) * 192;
      const half_t* src = Kp + rowbase + (size_t)(k0 + r) * E_TOTAL + c0;
      half_t* dst = KV + r * KS_STRIDE + c0;
#if USE_ASYNC
      #pragma unroll 4
      for (int i = 0; i < 192; i += 8) async_copy16(dst + i, src + i);
#else
      #pragma unroll
      for (int i = 0; i < 192; i += 8) *(v8h*)(dst + i) = *(const v8h*)(src + i);
#endif
    }
#if USE_ASYNC
    ASYNC_WAIT(0);     // this wave's Q (first iter) + K copies have landed
#endif
    __syncthreads();

    // ---- S = Q @ K^T : wave handles one 16x16 tile of the 32x64 score block
    const int tr = wave >> 2;        // 0..1 query row block
    const int tc = wave & 3;         // 0..3 key col block
    v8f s = {};
    {
      const half_t* qrow = Qs + (tr * 16 + l16) * QS_STRIDE;
      const half_t* krow = KV + (tc * 16 + l16) * KS_STRIDE;
      #pragma unroll 4
      for (int kk = 0; kk < D_MODEL; kk += 32) {
        union { v16h v; v8h h[2]; } af, bf;
        af.h[0] = *(const v8h*)(qrow + kk + 16 * lhi);
        af.h[1] = *(const v8h*)(qrow + kk + 16 * lhi + 8);
        bf.h[0] = *(const v8h*)(krow + kk + 16 * lhi);
        bf.h[1] = *(const v8h*)(krow + kk + 16 * lhi + 8);
        s = __builtin_amdgcn_wmma_f32_16x16x32_f16(
                false, af.v, false, bf.v, (short)0, s, false, false);
      }
    }
    // causal mask + spill scores to LDS
    {
      const int mbase = tr * 16 + 8 * lhi;
      const int nloc  = tc * 16 + l16;
      const int kg    = k0 + nloc;
      #pragma unroll
      for (int r = 0; r < 8; ++r) {
        float v = s[r];
        if (kg > q0 + mbase + r) v = -3.0e38f;
        Ss[(mbase + r) * SS_STRIDE + nloc] = v;
      }
    }
    __syncthreads();   // K reads done, S complete

    // ---- stream V tile into KV buffer TRANSPOSED: Vt[d][key] ----
    // (stays on the VGPR path: transpose happens between load and ds_store)
    {
      const int r  = tid >> 2;               // key 0..63
      const int c0 = (tid & 3) * 192;        // d offset
      const half_t* src = Vp + rowbase + (size_t)(k0 + r) * E_TOTAL + c0;
      #pragma unroll 4
      for (int i = 0; i < 192; i += 8) {
        v8h v = *(const v8h*)(src + i);
        #pragma unroll
        for (int e = 0; e < 8; ++e)
          KV[(c0 + i + e) * VT_STRIDE + r] = v[e];
      }
    }
    // ---- online softmax: one thread per query row ----
    if (tid < MQ) {
      const int r = tid;
      float mold = mrow[r];
      float mnew = mold;
      for (int n = 0; n < NK; ++n) mnew = fmaxf(mnew, Ss[r * SS_STRIDE + n]);
      const float corr = __expf(mold - mnew);
      float lsum = 0.0f;
      for (int n = 0; n < NK; ++n) {
        float p = __expf(Ss[r * SS_STRIDE + n] - mnew);
        lsum += p;
        Ps[r * PS_STRIDE + n] = (half_t)p;
      }
      mrow[r] = mnew;
      lrow[r] = lrow[r] * corr + lsum;
      crow[r] = corr;
    }
    __syncthreads();   // Ps, Vt, stats ready

    // ---- rescale O by correction factor ----
    #pragma unroll
    for (int rb = 0; rb < 2; ++rb) {
      #pragma unroll
      for (int r = 0; r < 8; ++r) {
        const float c = crow[rb * 16 + 8 * lhi + r];
        #pragma unroll
        for (int cb = 0; cb < 6; ++cb) o[rb][cb][r] *= c;
      }
    }
    // ---- O += P @ V : wave owns D slice [96*wave, 96*wave+96) ----
    #pragma unroll
    for (int ks = 0; ks < 2; ++ks) {
      union { v16h v; v8h h[2]; } af[2];
      #pragma unroll
      for (int rb = 0; rb < 2; ++rb) {
        const half_t* p = Ps + (rb * 16 + l16) * PS_STRIDE + ks * 32;
        af[rb].h[0] = *(const v8h*)(p + 16 * lhi);
        af[rb].h[1] = *(const v8h*)(p + 16 * lhi + 8);
      }
      #pragma unroll
      for (int cb = 0; cb < 6; ++cb) {
        union { v16h v; v8h h[2]; } bf;
        const half_t* vp = KV + (wave * 96 + cb * 16 + l16) * VT_STRIDE + ks * 32 + 16 * lhi;
        bf.h[0] = *(const v8h*)vp;
        bf.h[1] = *(const v8h*)(vp + 8);
        #pragma unroll
        for (int rb = 0; rb < 2; ++rb)
          o[rb][cb] = __builtin_amdgcn_wmma_f32_16x16x32_f16(
              false, af[rb].v, false, bf.v, (short)0, o[rb][cb], false, false);
      }
    }
  }
  __syncthreads();

  // ---- epilogue: Out = X + O / l ----
  #pragma unroll
  for (int rb = 0; rb < 2; ++rb) {
    #pragma unroll
    for (int cb = 0; cb < 6; ++cb) {
      const int mloc = rb * 16 + 8 * lhi;
      const int col  = wave * 96 + cb * 16 + l16;
      const size_t base = ((size_t)(b * N_CTX + q0 + mloc)) * D_MODEL + col;
      #pragma unroll
      for (int r = 0; r < 8; ++r) {
        const float rl = 1.0f / lrow[mloc + r];
        Out[base + (size_t)r * D_MODEL] = X[base + (size_t)r * D_MODEL] + o[rb][cb][r] * rl;
      }
    }
  }
}

// ---------------------------------------------------------------------------
// Host launcher
// ---------------------------------------------------------------------------
extern "C" void kernel_launch(void* const* d_in, const int* in_sizes, int n_in,
                              void* d_out, int out_size, void* d_ws, size_t ws_size,
                              hipStream_t stream) {
  (void)in_sizes; (void)n_in; (void)out_size; (void)ws_size;
  const float* x = (const float*)d_in[0];          // [2][4096][768] f32
  const float* w = (const float*)d_in[1];          // [2304][768]    f32
  float* out = (float*)d_out;                      // [2][4096][768] f32

  char* ws = (char*)d_ws;
  const size_t xh_bytes = (size_t)M_TOTAL * D_MODEL * 2;   // 12.6 MB
  const size_t wh_bytes = (size_t)E_TOTAL * D_MODEL * 2;   //  3.5 MB
  half_t* Xh   = (half_t*)ws;
  half_t* Wh   = (half_t*)(ws + xh_bytes);
  half_t* QKVh = (half_t*)(ws + xh_bytes + wh_bytes);      // 37.7 MB

  const int nx = M_TOTAL * D_MODEL;   // 6291456
  const int nw = E_TOTAL * D_MODEL;   // 1769472
  cvt_f32_f16<<<nx / 4 / 256, 256, 0, stream>>>(x, Xh, nx);
  cvt_f32_f16<<<nw / 4 / 256, 256, 0, stream>>>(w, Wh, nw);

  dim3 g1(M_TOTAL / GBM, E_TOTAL / GBN);  // 64 x 18
  qkv_gemm<<<g1, 256, 0, stream>>>(Xh, Wh, QKVh);

  // ~170 KB dynamic LDS (<= 320 KB/WGP on CDNA5)
  (void)hipFuncSetAttribute(reinterpret_cast<const void*>(attn_kernel),
                            hipFuncAttributeMaxDynamicSharedMemorySize, ATTN_LDS_BYTES);
  attn_kernel<<<BATCH * (N_CTX / MQ), 256, ATTN_LDS_BYTES, stream>>>(QKVh, x, out);
}
